// AttentionDecoder_22196390986438
// MI455X (gfx1250) — matrix operand order repaired and checked by reading, not compiled, not run on hardware
//
#include <hip/hip_runtime.h>
#include <hip/hip_bf16.h>
#include <math.h>

// Set to 0 if the gfx1250 async-to-LDS asm fails to assemble.
#define USE_ASYNC_LDS 1

// ---------------------------------------------------------------------------
// Types for CDNA5 WMMA (wave32): v16bf = 16 x bf16 per lane = one 16x32 (A) or
// 32x16 (B) bf16 operand; v8f = 16x16 f32 accumulator tile (8 VGPRs/lane).
// ---------------------------------------------------------------------------
typedef __bf16 bhalf;
typedef bhalf        v16bf __attribute__((ext_vector_type(16)));
typedef float        v8f   __attribute__((ext_vector_type(8)));
typedef unsigned int v8u   __attribute__((ext_vector_type(8)));

__device__ __forceinline__ unsigned short f32_to_bf16(float f) {
  unsigned int u = __float_as_uint(f);
  u += 0x7FFFu + ((u >> 16) & 1u);   // round-to-nearest-even
  return (unsigned short)(u >> 16);
}

// ---------------------------------------------------------------------------
// fp32 -> bf16 conversion (grid-stride), used for encoder hiddens.
// ---------------------------------------------------------------------------
__global__ void cvt_f32_bf16(const float* __restrict__ src,
                             unsigned short* __restrict__ dst, long n) {
  long i = (long)blockIdx.x * blockDim.x + threadIdx.x;
  long stride = (long)gridDim.x * blockDim.x;
  for (; i < n; i += stride) dst[i] = f32_to_bf16(src[i]);
}

// ---------------------------------------------------------------------------
// Tiled transpose + fp32->bf16: src is [R, C] row-major fp32,
// dst is [C, R] row-major bf16. Block (32, 8), 32x32 tile via LDS.
// ---------------------------------------------------------------------------
__global__ __launch_bounds__(256) void transpose_cvt(
    const float* __restrict__ src, unsigned short* __restrict__ dst,
    int R, int C) {
  __shared__ float tile[32][33];
  int c0 = blockIdx.x * 32, r0 = blockIdx.y * 32;
  int tx = threadIdx.x, ty = threadIdx.y;
#pragma unroll
  for (int i = 0; i < 4; ++i)
    tile[ty + i * 8][tx] = src[(long)(r0 + ty + i * 8) * C + c0 + tx];
  __syncthreads();
#pragma unroll
  for (int i = 0; i < 4; ++i)
    dst[(long)(c0 + ty + i * 8) * R + r0 + tx] = f32_to_bf16(tile[tx][ty + i * 8]);
}

// ---------------------------------------------------------------------------
// Embedding lookup + sinusoidal positional encoding. c = -ln(10000)/D.
// ---------------------------------------------------------------------------
__global__ void embed_pe(const int* __restrict__ trg, const float* __restrict__ emb,
                         float* __restrict__ xf, unsigned short* __restrict__ xb,
                         int T, int D, float c) {
  long i = (long)blockIdx.x * blockDim.x + threadIdx.x;
  int d = (int)(i % D);
  long bt = i / D;
  int tpos = (int)(bt % T);
  int tok = trg[bt];
  int half = d >> 1;
  float freq = __expf((float)(2 * half) * c);
  float ang = (float)tpos * freq;
  float pe = (d & 1) ? __cosf(ang) : __sinf(ang);
  float val = emb[(long)tok * D + d] + pe;
  xf[i] = val;
  xb[i] = f32_to_bf16(val);
}

// ---------------------------------------------------------------------------
// bf16 WMMA GEMM:  C[M,N] = A[M,K] @ B[K,N] (+bias) (ReLU optional)
// A row-major bf16 [M,K]. B supplied TRANSPOSED: Bt[n*ldb + k] (contiguous K),
// so both tiles stage as aligned b128 transfers. Batched via blockIdx.z.
// M,N multiples of 128; K multiple of 32; lda/ldb multiples of 8.
// Block = 256 threads = 8 waves; wave (wm 0..3, wn 0..1) owns a 32x64 subtile
// = 2x4 v_wmma_f32_16x16x32_bf16 accumulators. Double-buffered LDS with
// gfx1250 global_load_async_to_lds_b128 prefetch (ASYNCcnt).
// storeT: write bf16 output transposed: Cb[n*ldct + m] (used to produce V^T).
// ---------------------------------------------------------------------------
#define BM 128
#define BN 128
#define BK 32
#define LROW 20  // dwords per LDS tile row: 16 data + 4 pad (16B aligned,
                 // 20*L mod 64 distinct for L=0..15 -> conflict-free reads)

__global__ __launch_bounds__(256) void wmma_gemm_bf16(
    const unsigned short* __restrict__ A, int lda, long strideA,
    const unsigned short* __restrict__ Bt, int ldb, long strideB,
    const float* __restrict__ bias,
    float* __restrict__ Cf, unsigned short* __restrict__ Cb, int ldc, long strideC,
    int K, int relu, int storeT, int ldct) {
  __shared__ unsigned int Asb[2][BM * LROW];
  __shared__ unsigned int Bsb[2][BN * LROW];

  const int t = threadIdx.x;
  const int lane = t & 31;
  const int w = t >> 5;
  const int waveM = w >> 1;   // 0..3
  const int waveN = w & 1;    // 0..1
  const int lrow = lane & 15;
  const int lhalf = lane >> 4;

  const int m0 = blockIdx.y * BM;
  const int n0 = blockIdx.x * BN;
  const int bz = blockIdx.z;

  const unsigned short* Ap = A + (long)bz * strideA;
  const unsigned short* Bp = Bt + (long)bz * strideB;

  // thread's slice of the staging work: 2 x b128 per tile for A, 2 for B
  const int srow0 = t >> 2;            // 0..63   (i=0)
  const int srow1 = 64 + (t >> 2);     // 64..127 (i=1)
  const int sq = t & 3;                // uint4 column within row

  auto issue_tile = [&](int k0, int buf) {
#if USE_ASYNC_LDS
#pragma unroll
    for (int i = 0; i < 2; ++i) {
      int row = (i == 0) ? srow0 : srow1;
      unsigned la = (unsigned)(size_t)&Asb[buf][row * LROW + sq * 4];
      const unsigned short* ga = Ap + (long)(m0 + row) * lda + k0 + sq * 8;
      asm volatile("global_load_async_to_lds_b128 %0, %1, off"
                   :: "v"(la), "v"(ga) : "memory");
      unsigned lb = (unsigned)(size_t)&Bsb[buf][row * LROW + sq * 4];
      const unsigned short* gb = Bp + (long)(n0 + row) * ldb + k0 + sq * 8;
      asm volatile("global_load_async_to_lds_b128 %0, %1, off"
                   :: "v"(lb), "v"(gb) : "memory");
    }
#else
    uint4 va[2], vb[2];
#pragma unroll
    for (int i = 0; i < 2; ++i) {
      int row = (i == 0) ? srow0 : srow1;
      va[i] = *(const uint4*)(Ap + (long)(m0 + row) * lda + k0 + sq * 8);
      vb[i] = *(const uint4*)(Bp + (long)(n0 + row) * ldb + k0 + sq * 8);
    }
#pragma unroll
    for (int i = 0; i < 2; ++i) {
      int row = (i == 0) ? srow0 : srow1;
      *(uint4*)&Asb[buf][row * LROW + sq * 4] = va[i];
      *(uint4*)&Bsb[buf][row * LROW + sq * 4] = vb[i];
    }
#endif
  };

  v8f acc[2][4];
  v8f zero = {0.f, 0.f, 0.f, 0.f, 0.f, 0.f, 0.f, 0.f};
#pragma unroll
  for (int s = 0; s < 2; ++s)
#pragma unroll
    for (int nt = 0; nt < 4; ++nt) acc[s][nt] = zero;

  const int nTiles = K / BK;
  issue_tile(0, 0);

  for (int it = 0; it < nTiles; ++it) {
#if USE_ASYNC_LDS
    asm volatile("s_wait_asynccnt 0x0" ::: "memory");
#endif
    __syncthreads();  // tile `it` resident; everyone done reading buf (it+1)&1
    if (it + 1 < nTiles) issue_tile((it + 1) * BK, (it + 1) & 1);

    const unsigned int* As = Asb[it & 1];
    const unsigned int* Bs = Bsb[it & 1];

    v8u au[2], bu[4];
#pragma unroll
    for (int s = 0; s < 2; ++s) {
      int row = waveM * 32 + s * 16 + lrow;
#pragma unroll
      for (int v = 0; v < 8; ++v) {
        // A 16x32 bf16: VGPR v holds K pair (v&3)*2 + lhalf*8 + (v>>2)*16
        int kp = (v & 3) + lhalf * 4 + (v >> 2) * 8;
        au[s][v] = As[row * LROW + kp];
      }
    }
#pragma unroll
    for (int nt = 0; nt < 4; ++nt) {
      int coln = waveN * 64 + nt * 16 + lrow;
#pragma unroll
      for (int v = 0; v < 8; ++v) {
        // B 32x16 bf16: VGPR v holds K pair (v&3)*2 + lhalf*8 + (v>>2)*16
        int kp = (v & 3) + lhalf * 4 + (v >> 2) * 8;
        bu[nt][v] = Bs[coln * LROW + kp];
      }
    }
#pragma unroll
    for (int s = 0; s < 2; ++s) {
      v16bf af = __builtin_bit_cast(v16bf, au[s]);
#pragma unroll
      for (int nt = 0; nt < 4; ++nt) {
        v16bf bf = __builtin_bit_cast(v16bf, bu[nt]);
        acc[s][nt] = __builtin_amdgcn_wmma_f32_16x16x32_bf16(
            false, af, false, bf, (short)0, acc[s][nt], false, false);
      }
    }
  }

  // ---- epilogue: C/D layout row = r + 8*lhalf, col = lane&15 ----
  float* Cfp = Cf ? Cf + (long)bz * strideC : nullptr;
  unsigned short* Cbp = Cb ? Cb + (long)bz * strideC : nullptr;
#pragma unroll
  for (int s = 0; s < 2; ++s) {
#pragma unroll
    for (int nt = 0; nt < 4; ++nt) {
      int col = n0 + waveN * 64 + nt * 16 + lrow;
      float bv = bias ? bias[col] : 0.f;
#pragma unroll
      for (int r = 0; r < 8; ++r) {
        int row = m0 + waveM * 32 + s * 16 + lhalf * 8 + r;
        float vv = acc[s][nt][r] + bv;
        if (relu) vv = fmaxf(vv, 0.f);
        if (storeT) {
          Cbp[(long)col * ldct + row] = f32_to_bf16(vv);
        } else {
          long o = (long)row * ldc + col;
          if (Cfp) Cfp[o] = vv;
          if (Cbp) Cbp[o] = f32_to_bf16(vv);
        }
      }
    }
  }
}

// ---------------------------------------------------------------------------
// Fused scale + mask + softmax over 512 columns; one block per (b, q) row.
// mode 0: self-attn (causal AND key-token != 0); mode 1: cross (k < srclen[b]).
// ---------------------------------------------------------------------------
__global__ __launch_bounds__(256) void softmax_mask(
    const float* __restrict__ Sc, unsigned short* __restrict__ Pb,
    const int* __restrict__ trg, const int* __restrict__ srclen,
    int T, int NC, float scale, int mode) {
  int row = blockIdx.x;  // b*T + q
  int b = row / T, q = row % T;
  int t = threadIdx.x;
  const float* srow = Sc + (long)row * NC;
  unsigned short* prow = Pb + (long)row * NC;
  __shared__ float red[256];
  float vals[2];
  float lmax = -3.4e38f;
#pragma unroll
  for (int i = 0; i < 2; ++i) {
    int k = t + i * 256;
    bool ok = (mode == 0) ? ((trg[b * T + k] != 0) && (k <= q)) : (k < srclen[b]);
    float v = ok ? srow[k] * scale : -1e9f;
    vals[i] = v;
    lmax = fmaxf(lmax, v);
  }
  red[t] = lmax;
  __syncthreads();
  for (int s = 128; s > 0; s >>= 1) {
    if (t < s) red[t] = fmaxf(red[t], red[t + s]);
    __syncthreads();
  }
  float m = red[0];
  __syncthreads();
  float lsum = 0.f;
#pragma unroll
  for (int i = 0; i < 2; ++i) {
    vals[i] = __expf(vals[i] - m);
    lsum += vals[i];
  }
  red[t] = lsum;
  __syncthreads();
  for (int s = 128; s > 0; s >>= 1) {
    if (t < s) red[t] += red[t + s];
    __syncthreads();
  }
  float inv = 1.f / red[0];
#pragma unroll
  for (int i = 0; i < 2; ++i) prow[t + i * 256] = f32_to_bf16(vals[i] * inv);
}

// ---------------------------------------------------------------------------
// Fused residual add + LayerNorm over D=512; one block per row.
// ---------------------------------------------------------------------------
__global__ __launch_bounds__(256) void add_layernorm(
    const float* __restrict__ X, const float* __restrict__ R,
    const float* __restrict__ g, const float* __restrict__ bta,
    float* __restrict__ Yf, unsigned short* __restrict__ Yb, int D) {
  int row = blockIdx.x;
  int t = threadIdx.x;
  const float* x = X + (long)row * D;
  const float* r = R + (long)row * D;
  __shared__ float rs[256], rs2[256];
  float v[2];
  float s = 0.f, s2 = 0.f;
#pragma unroll
  for (int i = 0; i < 2; ++i) {
    int d = t + i * 256;
    float z = x[d] + r[d];
    v[i] = z;
    s += z;
    s2 += z * z;
  }
  rs[t] = s;
  rs2[t] = s2;
  __syncthreads();
  for (int k = 128; k > 0; k >>= 1) {
    if (t < k) { rs[t] += rs[t + k]; rs2[t] += rs2[t + k]; }
    __syncthreads();
  }
  float mean = rs[0] / D;
  float var = rs2[0] / D - mean * mean;
  float rstd = rsqrtf(var + 1e-5f);
#pragma unroll
  for (int i = 0; i < 2; ++i) {
    int d = t + i * 256;
    float y = (v[i] - mean) * rstd * g[d] + bta[d];
    if (Yf) Yf[(long)row * D + d] = y;
    Yb[(long)row * D + d] = f32_to_bf16(y);
  }
}

// ---------------------------------------------------------------------------
extern "C" void kernel_launch(void* const* d_in, const int* in_sizes, int n_in,
                              void* d_out, int out_size, void* d_ws, size_t ws_size,
                              hipStream_t stream) {
  (void)in_sizes; (void)n_in; (void)out_size; (void)ws_size;
  const int B = 8, T = 512, S = 512, D = 512, V = 32000, DF = 2048;

  const int*   trg  = (const int*)d_in[0];
  const float* enc  = (const float*)d_in[1];
  const int*   srcl = (const int*)d_in[2];
  const float* emb  = (const float*)d_in[3];
  const float* wq1 = (const float*)d_in[4];  const float* bq1 = (const float*)d_in[5];
  const float* wk1 = (const float*)d_in[6];  const float* bk1 = (const float*)d_in[7];
  const float* wv1 = (const float*)d_in[8];  const float* bv1 = (const float*)d_in[9];
  const float* wq2 = (const float*)d_in[10]; const float* bq2 = (const float*)d_in[11];
  const float* wk2 = (const float*)d_in[12]; const float* bk2 = (const float*)d_in[13];
  const float* wv2 = (const float*)d_in[14]; const float* bv2 = (const float*)d_in[15];
  const float* w1  = (const float*)d_in[16]; const float* b1  = (const float*)d_in[17];
  const float* w2  = (const float*)d_in[18]; const float* b2  = (const float*)d_in[19];
  const float* fcw = (const float*)d_in[20]; const float* fcb = (const float*)d_in[21];
  const float* g1  = (const float*)d_in[22]; const float* be1 = (const float*)d_in[23];
  const float* g2  = (const float*)d_in[24]; const float* be2 = (const float*)d_in[25];
  const float* g3  = (const float*)d_in[26]; const float* be3 = (const float*)d_in[27];

  char* ws = (char*)d_ws;
  size_t off = 0;
  auto alloc = [&](size_t bytes) -> void* {
    void* p = ws + off;
    off += (bytes + 255) & ~(size_t)255;
    return p;
  };
  typedef unsigned short u16;
  // transposed bf16 weights: Wt[n*K + k]
  u16* wq1t = (u16*)alloc((size_t)D * D * 2);
  u16* wk1t = (u16*)alloc((size_t)D * D * 2);
  u16* wv1t = (u16*)alloc((size_t)D * D * 2);
  u16* wq2t = (u16*)alloc((size_t)D * D * 2);
  u16* wk2t = (u16*)alloc((size_t)D * D * 2);
  u16* wv2t = (u16*)alloc((size_t)D * D * 2);
  u16* w1t  = (u16*)alloc((size_t)D * DF * 2);
  u16* w2t  = (u16*)alloc((size_t)DF * D * 2);
  u16* fcwt = (u16*)alloc((size_t)D * V * 2);
  u16* encb = (u16*)alloc((size_t)B * S * D * 2);
  float* xf   = (float*)alloc((size_t)B * T * D * 4);
  u16*   xb   = (u16*)alloc((size_t)B * T * D * 2);
  u16*   qb   = (u16*)alloc((size_t)B * T * D * 2);
  u16*   kb   = (u16*)alloc((size_t)B * S * D * 2);
  u16*   vtb  = (u16*)alloc((size_t)B * D * S * 2);   // V^T per batch [D, S]
  float* scf  = (float*)alloc((size_t)B * T * S * 4);
  u16*   scb  = (u16*)alloc((size_t)B * T * S * 2);
  float* tmpf = (float*)alloc((size_t)B * T * D * 4);
  float* x2f  = (float*)alloc((size_t)B * T * D * 4);
  u16*   x2b  = (u16*)alloc((size_t)B * T * D * 2);
  float* x3f  = (float*)alloc((size_t)B * T * D * 4);
  u16*   x3b  = (u16*)alloc((size_t)B * T * D * 2);
  u16*   hb   = (u16*)alloc((size_t)B * T * DF * 2);
  u16*   x4b  = (u16*)alloc((size_t)B * T * D * 2);

  auto tcvt = [&](const float* s, u16* d, int R, int C) {
    transpose_cvt<<<dim3(C / 32, R / 32), dim3(32, 8), 0, stream>>>(s, d, R, C);
  };
  tcvt(wq1, wq1t, D, D); tcvt(wk1, wk1t, D, D); tcvt(wv1, wv1t, D, D);
  tcvt(wq2, wq2t, D, D); tcvt(wk2, wk2t, D, D); tcvt(wv2, wv2t, D, D);
  tcvt(w1, w1t, D, DF);  tcvt(w2, w2t, DF, D);  tcvt(fcw, fcwt, D, V);
  cvt_f32_bf16<<<2048, 256, 0, stream>>>(enc, encb, (long)B * S * D);

  embed_pe<<<(B * T * D) / 256, 256, 0, stream>>>(trg, emb, xf, xb, T, D,
                                                  -logf(10000.0f) / (float)D);

  auto gemm = [&](const u16* A, int lda, long sA, const u16* Bt, int ldb, long sB,
                  const float* bias, float* Cf, u16* Cb, int ldc, long sC,
                  int M, int N, int K, int batch, int relu, int storeT, int ldct) {
    dim3 grid(N / 128, M / 128, batch);
    wmma_gemm_bf16<<<grid, 256, 0, stream>>>(A, lda, sA, Bt, ldb, sB, bias,
                                             Cf, Cb, ldc, sC, K, relu, storeT, ldct);
  };
  const float scale = 1.0f / sqrtf((float)D);

  // ---- self-attention ----
  gemm(xb, D, 0, wq1t, D, 0, bq1, nullptr, qb, D, 0, B * T, D, D, 1, 0, 0, 0);
  gemm(xb, D, 0, wk1t, D, 0, bk1, nullptr, kb, D, 0, B * T, D, D, 1, 0, 0, 0);
  gemm(xb, D, (long)T * D, wv1t, D, 0, bv1, nullptr, vtb, 0, (long)D * T,
       T, D, D, B, 0, 1, T);                                        // -> V^T
  gemm(qb, D, (long)T * D, kb, D, (long)T * D, nullptr, scf, nullptr,
       T, (long)T * T, T, T, D, B, 0, 0, 0);                        // Q @ K^T
  softmax_mask<<<B * T, 256, 0, stream>>>(scf, scb, trg, srcl, T, T, scale, 0);
  gemm(scb, T, (long)T * T, vtb, T, (long)D * T, nullptr, tmpf, nullptr,
       D, (long)T * D, T, D, T, B, 0, 0, 0);                        // P @ V
  add_layernorm<<<B * T, 256, 0, stream>>>(xf, tmpf, g1, be1, x2f, x2b, D);

  // ---- cross-attention ----
  gemm(x2b, D, 0, wq2t, D, 0, bq2, nullptr, qb, D, 0, B * T, D, D, 1, 0, 0, 0);
  gemm(encb, D, 0, wk2t, D, 0, bk2, nullptr, kb, D, 0, B * S, D, D, 1, 0, 0, 0);
  gemm(encb, D, (long)S * D, wv2t, D, 0, bv2, nullptr, vtb, 0, (long)D * S,
       S, D, D, B, 0, 1, S);                                        // -> V^T
  gemm(qb, D, (long)T * D, kb, D, (long)S * D, nullptr, scf, nullptr,
       S, (long)T * S, T, S, D, B, 0, 0, 0);
  softmax_mask<<<B * T, 256, 0, stream>>>(scf, scb, trg, srcl, T, S, scale, 1);
  gemm(scb, S, (long)T * S, vtb, S, (long)D * S, nullptr, tmpf, nullptr,
       D, (long)T * D, T, D, S, B, 0, 0, 0);
  add_layernorm<<<B * T, 256, 0, stream>>>(x2f, tmpf, g2, be2, x3f, x3b, D);

  // ---- FFN ----
  gemm(x3b, D, 0, w1t, D, 0, b1, nullptr, hb, DF, 0, B * T, DF, D, 1, 1, 0, 0);
  gemm(hb, DF, 0, w2t, DF, 0, b2, tmpf, nullptr, D, 0, B * T, D, DF, 1, 0, 0, 0);
  add_layernorm<<<B * T, 256, 0, stream>>>(x3f, tmpf, g3, be3, nullptr, x4b, D);

  // ---- vocab projection (writes d_out fp32 directly) ----
  gemm(x4b, D, 0, fcwt, D, 0, fcb, (float*)d_out, nullptr, V, 0,
       B * T, V, D, 1, 0, 0, 0);
}
